// VQVAE_33397665694421
// MI455X (gfx1250) — matrix operand order repaired and checked by reading, not compiled, not run on hardware
//
#include <hip/hip_runtime.h>

// CDNA5 / gfx1250 wave32 WMMA implementation of a VQ-VAE-2 forward pass.
// All convs (incl. transposed) run as implicit GEMM via v_wmma_f32_16x16x32_f16.
// Block tile 32(M) x 128(N); 8 waves (2x4); each wave: 16x32 tile = 2 WMMAs/K-step.
// Kernel templated on (KHW, KW, TRANS) so all gather divisions strength-reduce.
// Workspace requirement: ~80 MB (bump-allocated below).

typedef __attribute__((ext_vector_type(16))) _Float16 v16h;
typedef __attribute__((ext_vector_type(8)))  float    v8f;

#define F_INRELU  1
#define F_OUTRELU 2
#define F_TRANS   4

#define ASTR 48   // LDS row stride (halves) for A tile (32x32): 96B rows, 16B-aligned chunks
#define BSTR 48   // LDS row stride (halves) for B tile stored n-major (128 x 32)

// ---------------------------------------------------------------------------
// KHW = kh*kw (compile-time), KW = kw (compile-time), TR = transposed (stride 2).
// ---------------------------------------------------------------------------
template <int KHW, int KW, bool TR>
__global__ __launch_bounds__(256) void conv_wmma(
    const _Float16* __restrict__ A, const float* __restrict__ bias,
    const _Float16* __restrict__ X,
    _Float16* __restrict__ outH, float* __restrict__ outF,
    const _Float16* __restrict__ resid,
    int Nb, int Cin, int IH, int IW,
    int Cout, int OH, int OW,
    int stride, int pad,
    int Kp, int Ctot, int cofs, int flags)
{
  __shared__ _Float16 As[32 * ASTR];
  __shared__ _Float16 Bs[128 * BSTR];

  const int tid  = threadIdx.x;
  const int lane = tid & 31;
  const int wave = tid >> 5;
  const int wm   = (wave >> 2) << 4;   // 0 or 16
  const int wn   = (wave & 3) << 5;    // 0,32,64,96
  const int p0   = blockIdx.x * 128;   // pixel tile
  const int m0   = blockIdx.y * 32;    // cout tile
  const int P    = Nb * OH * OW;
  const int Kr   = Cin * KHW;
  const int ohw  = OH * OW;
  const bool inRelu = (flags & F_INRELU) != 0;

  // ---- per-thread B-gather pixel (fixed across the K loop) ----
  const int gnl  = tid >> 1;             // 0..127 (pixel within tile)
  const int klb  = (tid & 1) << 4;       // 0 or 16 (K sub-chunk)
  const int gp   = p0 + gnl;
  const bool pok = gp < P;
  int gn = 0, goy = 0, gox = 0;
  if (pok) {
    gn  = gp / ohw; int pr = gp - gn * ohw;
    goy = pr / OW;  gox = pr - goy * OW;
  }
  const size_t xbase = (size_t)gn * Cin * IH * IW;

  v8f acc0 = {}, acc1 = {};

  for (int k0 = 0; k0 < Kp; k0 += 32) {
    // ---- stage A tile (32 x 32 halves), 4 contiguous halves per thread ----
    {
      int m  = tid >> 3;
      int kk = (tid & 7) << 2;
      int gm = m0 + m;
      _Float16 v0 = 0, v1 = 0, v2 = 0, v3 = 0;
      if (gm < Cout) {
        const _Float16* s = A + (size_t)gm * Kp + k0 + kk;
        v0 = s[0]; v1 = s[1]; v2 = s[2]; v3 = s[3];
      }
      _Float16* d = &As[m * ASTR + kk];
      d[0] = v0; d[1] = v1; d[2] = v2; d[3] = v3;
    }
    // ---- stage B tile (im2col gather), n-major: Bs[n][k], 16 halves/thread ----
    {
      _Float16* d = &Bs[gnl * BSTR + klb];
      #pragma unroll
      for (int i = 0; i < 16; ++i) {
        int k = k0 + klb + i;
        float v = 0.f;
        if (pok && k < Kr) {
          int ci = k / KHW;             // compile-time divisor -> mul/shift
          int r  = k - ci * KHW;
          int ky = r / KW;
          int kx = r - ky * KW;
          int iy, ix; bool ok;
          if (!TR) {
            iy = goy * stride - pad + ky;
            ix = gox * stride - pad + kx;
            ok = (iy >= 0) & (iy < IH) & (ix >= 0) & (ix < IW);
          } else {
            int ty = goy + pad - ky;    // stride fixed at 2 for transposed convs
            int tx = gox + pad - kx;
            ok = (ty >= 0) & (tx >= 0) & ((ty & 1) == 0) & ((tx & 1) == 0);
            iy = ty >> 1; ix = tx >> 1;
            ok = ok && (iy < IH) && (ix < IW);
          }
          if (ok) {
            float xv = (float)X[xbase + ((size_t)ci * IH + iy) * IW + ix];
            if (inRelu && xv < 0.f) xv = 0.f;
            v = xv;
          }
        }
        d[i] = (_Float16)v;
      }
    }
    __syncthreads();

    // ---- fragments per CDNA5 wave32 layouts ----
    const int hi = lane >> 4;           // K-half select
    const int ml = wm + (lane & 15);    // A row (M)
    const int n0 = wn + (lane & 15);    // B col (N), sub-tile 0
    v16h a, b0, b1;
    const _Float16* ap = &As[ml * ASTR + (hi << 4)];
    #pragma unroll
    for (int e = 0; e < 8; ++e) a[e]     = ap[e];        // K = hi*16 + e
    #pragma unroll
    for (int e = 0; e < 8; ++e) a[8 + e] = ap[8 + e];    // K = hi*16 + 8 + e
    const _Float16* bp0 = &Bs[n0 * BSTR + (hi << 4)];
    const _Float16* bp1 = bp0 + 16 * BSTR;               // N + 16
    #pragma unroll
    for (int e = 0; e < 16; ++e) b0[e] = bp0[e];         // K = hi*16 + e
    #pragma unroll
    for (int e = 0; e < 16; ++e) b1[e] = bp1[e];

    acc0 = __builtin_amdgcn_wmma_f32_16x16x32_f16(
        false, a, false, b0, (short)0, acc0, false, false);
    acc1 = __builtin_amdgcn_wmma_f32_16x16x32_f16(
        false, a, false, b1, (short)0, acc1, false, false);
    __syncthreads();
  }

  // ---- epilogue: bias, optional residual, optional relu, store NCHW ----
  const int hi = lane >> 4;
  #pragma unroll
  for (int s = 0; s < 2; ++s) {
    const v8f accv = s ? acc1 : acc0;
    int p = p0 + wn + (s << 4) + (lane & 15);
    if (p < P) {
      int n  = p / ohw;  int pr = p - n * ohw;
      int oy = pr / OW;  int ox = pr - oy * OW;
      #pragma unroll
      for (int r = 0; r < 8; ++r) {
        int m = m0 + wm + (hi << 3) + r;
        if (m < Cout) {
          float v = accv[r] + bias[m];
          if (resid)
            v += (float)resid[(((size_t)n * Cout + m) * OH + oy) * OW + ox];
          if ((flags & F_OUTRELU) && v < 0.f) v = 0.f;
          size_t oi = (((size_t)n * Ctot + cofs + m) * OH + oy) * OW + ox;
          if (outH) outH[oi] = (_Float16)v;
          if (outF) outF[oi] = v;
        }
      }
    }
  }
}

// ---------------------------------------------------------------------------
// Weight repack: fp32 (O,I,kh,kw) or transposed (I,O,kh,kw) -> fp16 [Cout][Kp]
// ---------------------------------------------------------------------------
__global__ void pack_w(const float* __restrict__ W, _Float16* __restrict__ A,
                       int Cout, int Cin, int khw, int Kp, int tr)
{
  int i = blockIdx.x * 256 + threadIdx.x;
  int total = Cout * Kp;
  if (i >= total) return;
  int co = i / Kp, k = i - co * Kp;
  float v = 0.f;
  if (k < Cin * khw) {
    int ci = k / khw, r = k - ci * khw;
    v = tr ? W[((size_t)ci * Cout + co) * khw + r]
           : W[((size_t)co * Cin + ci) * khw + r];
  }
  A[i] = (_Float16)v;
}

__global__ void f2h(const float* __restrict__ s, _Float16* __restrict__ d, int n)
{
  int i = blockIdx.x * 256 + threadIdx.x;
  if (i < n) d[i] = (_Float16)s[i];
}

// channel-offset copy (for concat), optional relu
__global__ void copy_ch(const _Float16* __restrict__ src, _Float16* __restrict__ dst,
                        int Nel, int C, int HW, int Ctot, int cofs, int doRelu)
{
  int i = blockIdx.x * 256 + threadIdx.x;
  if (i >= Nel) return;
  int chw = C * HW;
  int n = i / chw; int r = i - n * chw;
  int c = r / HW;  int pr = r - c * HW;
  float v = (float)src[i];
  if (doRelu && v < 0.f) v = 0.f;
  dst[((size_t)n * Ctot + cofs + c) * HW + pr] = (_Float16)v;
}

// ---------------------------------------------------------------------------
// Vector quantization: one wave per pixel, 512 codes / 32 lanes = 16 per lane.
// ---------------------------------------------------------------------------
__global__ __launch_bounds__(256) void vq_kernel(
    const _Float16* __restrict__ Z, const float* __restrict__ E,
    _Float16* __restrict__ Q, float* __restrict__ sse,
    int Npix, int HW, int W, int dim, int K, int Ctot, int cofs)
{
  __shared__ float zb[8][64];
  __shared__ float red[8];
  const int lane = threadIdx.x & 31;
  const int wave = threadIdx.x >> 5;
  const int p = blockIdx.x * 8 + wave;
  const bool active = p < Npix;
  int n = 0, pr = 0;
  if (active) {
    n = p / HW; pr = p - n * HW;
    size_t base = (size_t)n * dim * HW + pr;
    zb[wave][lane]      = (float)Z[base + (size_t)lane * HW];
    zb[wave][lane + 32] = (float)Z[base + (size_t)(lane + 32) * HW];
  }
  __syncthreads();

  float bestd = 3.4e38f; int bestk = 0x7fffffff;
  if (active) {
    for (int j = 0; j < 16; ++j) {
      int k = lane + (j << 5);
      float dot = 0.f, nrm = 0.f;
      for (int d = 0; d < 64; ++d) {
        float e = E[d * K + k];
        dot += zb[wave][d] * e;
        nrm += e * e;
      }
      float dist = nrm - 2.f * dot;
      if (dist < bestd) { bestd = dist; bestk = k; }
    }
  }
  for (int off = 16; off; off >>= 1) {
    float od = __shfl_xor(bestd, off, 32);
    int   ok = __shfl_xor(bestk, off, 32);
    if (od < bestd || (od == bestd && ok < bestk)) { bestd = od; bestk = ok; }
  }

  float lsum = 0.f;
  if (active) {
    size_t ob = ((size_t)n * Ctot + cofs) * HW + pr;
    float q0 = E[lane * K + bestk];
    float q1 = E[(lane + 32) * K + bestk];
    Q[ob + (size_t)lane * HW]        = (_Float16)q0;
    Q[ob + (size_t)(lane + 32) * HW] = (_Float16)q1;
    float e0 = zb[wave][lane] - q0, e1 = zb[wave][lane + 32] - q1;
    lsum = e0 * e0 + e1 * e1;
  }
  for (int off = 16; off; off >>= 1) lsum += __shfl_xor(lsum, off, 32);
  if (lane == 0) red[wave] = lsum;
  __syncthreads();
  if (threadIdx.x == 0) {
    float s = 0.f;
    for (int w = 0; w < 8; ++w) s += red[w];
    atomicAdd(sse, s);
  }
}

__global__ void recon_sse(const float* __restrict__ dec, const float* __restrict__ x,
                          float* __restrict__ sse, int Nel)
{
  __shared__ float red[256];
  int i = blockIdx.x * 256 + threadIdx.x;
  float v = 0.f;
  if (i < Nel) { float d = dec[i] - x[i]; v = d * d; }
  red[threadIdx.x] = v;
  __syncthreads();
  for (int s = 128; s; s >>= 1) {
    if (threadIdx.x < (unsigned)s) red[threadIdx.x] += red[threadIdx.x + s];
    __syncthreads();
  }
  if (threadIdx.x == 0) atomicAdd(sse, red[0]);
}

__global__ void zero3(float* p) { if (threadIdx.x < 3) p[threadIdx.x] = 0.f; }

// total = recon_mean + (1 + BETA) * (mean_t + mean_b), BETA = 0.9
__global__ void finalize_total(const float* __restrict__ loss, float* __restrict__ out,
                               float invNt, float invNb, float invNr)
{
  out[0] = loss[2] * invNr + 1.9f * (loss[0] * invNt + loss[1] * invNb);
}

// ---------------------------------------------------------------------------
extern "C" void kernel_launch(void* const* d_in, const int* in_sizes, int n_in,
                              void* d_out, int out_size, void* d_ws, size_t ws_size,
                              hipStream_t stream)
{
  (void)in_sizes; (void)n_in; (void)out_size; (void)ws_size;
  const float* xf = (const float*)d_in[0];
  char* ws = (char*)d_ws;
  size_t off = 0;
  auto alloc = [&](size_t bytes) -> void* {
    void* p = ws + off;
    off = (off + bytes + 255) & ~(size_t)255;
    return p;
  };

  const int NB = 8;
  float*     loss  = (float*)    alloc(16);
  _Float16*  xh    = (_Float16*) alloc((size_t)NB * 3   * 256 * 256 * 2);
  _Float16*  a1    = (_Float16*) alloc((size_t)NB * 64  * 128 * 128 * 2);
  _Float16*  a2    = (_Float16*) alloc((size_t)NB * 128 * 64  * 64  * 2);
  _Float16*  a3    = (_Float16*) alloc((size_t)NB * 128 * 64  * 64  * 2);
  _Float16*  rtmp  = (_Float16*) alloc((size_t)NB * 32  * 64  * 64  * 2);
  _Float16*  b1    = (_Float16*) alloc((size_t)NB * 64  * 32  * 32  * 2);
  _Float16*  b2    = (_Float16*) alloc((size_t)NB * 128 * 32  * 32  * 2);
  _Float16*  b3    = (_Float16*) alloc((size_t)NB * 128 * 32  * 32  * 2);
  _Float16*  rtmpt = (_Float16*) alloc((size_t)NB * 32  * 32  * 32  * 2);
  _Float16*  zt    = (_Float16*) alloc((size_t)NB * 64  * 32  * 32  * 2);
  _Float16*  qt    = (_Float16*) alloc((size_t)NB * 64  * 32  * 32  * 2);
  _Float16*  cat   = (_Float16*) alloc((size_t)NB * 192 * 64  * 64  * 2);
  _Float16*  zb    = (_Float16*) alloc((size_t)NB * 64  * 64  * 64  * 2);
  _Float16*  qcat  = (_Float16*) alloc((size_t)NB * 128 * 64  * 64  * 2);

  auto pack = [&](int wIdx, int Cout, int Cin, int khw, int Kp, int tr) -> _Float16* {
    _Float16* dst = (_Float16*)alloc((size_t)Cout * Kp * 2);
    int total = Cout * Kp;
    pack_w<<<(total + 255) / 256, 256, 0, stream>>>(
        (const float*)d_in[wIdx], dst, Cout, Cin, khw, Kp, tr);
    return dst;
  };

  auto conv = [&](const _Float16* A, int bIdx, const _Float16* X,
                  _Float16* oh, float* of, const _Float16* res,
                  int Cin, int IH, int IW, int Cout, int OH, int OW,
                  int kh, int kw, int stride, int pad, int Kp,
                  int Ctot, int cofs, int flags) {
    int P = NB * OH * OW;
    dim3 grid((P + 127) / 128, (Cout + 31) / 32);
    const float* bias = (const float*)d_in[bIdx];
    if (flags & F_TRANS) {
      conv_wmma<16, 4, true><<<grid, 256, 0, stream>>>(
          A, bias, X, oh, of, res, NB, Cin, IH, IW, Cout, OH, OW,
          stride, pad, Kp, Ctot, cofs, flags);
    } else if (kh == 1) {
      conv_wmma<1, 1, false><<<grid, 256, 0, stream>>>(
          A, bias, X, oh, of, res, NB, Cin, IH, IW, Cout, OH, OW,
          stride, pad, Kp, Ctot, cofs, flags);
    } else if (kh == 3) {
      conv_wmma<9, 3, false><<<grid, 256, 0, stream>>>(
          A, bias, X, oh, of, res, NB, Cin, IH, IW, Cout, OH, OW,
          stride, pad, Kp, Ctot, cofs, flags);
    } else {
      conv_wmma<16, 4, false><<<grid, 256, 0, stream>>>(
          A, bias, X, oh, of, res, NB, Cin, IH, IW, Cout, OH, OW,
          stride, pad, Kp, Ctot, cofs, flags);
    }
  };

  auto vq = [&](const _Float16* Z, int eIdx, _Float16* Q, float* ssep,
                int H, int Wd, int Ctot, int cofs) {
    int Npix = NB * H * Wd;
    vq_kernel<<<(Npix + 7) / 8, 256, 0, stream>>>(
        Z, (const float*)d_in[eIdx], Q, ssep, Npix, H * Wd, Wd, 64, 512, Ctot, cofs);
  };

  // ---- init ----
  zero3<<<1, 32, 0, stream>>>(loss);
  {
    int n = NB * 3 * 256 * 256;
    f2h<<<(n + 255) / 256, 256, 0, stream>>>(xf, xh, n);
  }

  // ---- pack all weights (input dict flat order) ----
  _Float16* pEb1  = pack(1,  64,  3,   16, 64,   0);
  _Float16* pEb2  = pack(3,  128, 64,  16, 1024, 0);
  _Float16* pEb3  = pack(5,  128, 128, 9,  1152, 0);
  _Float16* pRE0a = pack(7,  32,  128, 9,  1152, 0);
  _Float16* pRE0b = pack(9,  128, 32,  1,  32,   0);
  _Float16* pRE1a = pack(11, 32,  128, 9,  1152, 0);
  _Float16* pRE1b = pack(13, 128, 32,  1,  32,   0);
  _Float16* pT1   = pack(15, 64,  128, 16, 2048, 0);
  _Float16* pT2   = pack(17, 128, 64,  9,  576,  0);
  _Float16* pRT0a = pack(19, 32,  128, 9,  1152, 0);
  _Float16* pRT0b = pack(21, 128, 32,  1,  32,   0);
  _Float16* pRT1a = pack(23, 32,  128, 9,  1152, 0);
  _Float16* pRT1b = pack(25, 128, 32,  1,  32,   0);
  _Float16* pTop  = pack(27, 64,  128, 1,  128,  0);
  _Float16* pDt1  = pack(30, 128, 64,  9,  576,  0);
  _Float16* pRD0a = pack(32, 32,  128, 9,  1152, 0);
  _Float16* pRD0b = pack(34, 128, 32,  1,  32,   0);
  _Float16* pRD1a = pack(36, 32,  128, 9,  1152, 0);
  _Float16* pRD1b = pack(38, 128, 32,  1,  32,   0);
  _Float16* pDtT  = pack(40, 64,  128, 16, 2048, 1);  // convT (Cin,Cout,4,4)
  _Float16* pBot  = pack(42, 64,  192, 1,  192,  0);
  _Float16* pUp   = pack(45, 64,  64,  16, 1024, 1);  // convT
  _Float16* pD1   = pack(47, 128, 128, 9,  1152, 0);
  _Float16* pRX0a = pack(49, 32,  128, 9,  1152, 0);
  _Float16* pRX0b = pack(51, 128, 32,  1,  32,   0);
  _Float16* pRX1a = pack(53, 32,  128, 9,  1152, 0);
  _Float16* pRX1b = pack(55, 128, 32,  1,  32,   0);
  _Float16* pDT1  = pack(57, 64,  128, 16, 2048, 1);  // convT
  _Float16* pDT2  = pack(59, 3,   64,  16, 1024, 1);  // convT

  // ---- encoder bottom: 256 -> 64 ----
  conv(pEb1, 2,  xh, a1, nullptr, nullptr, 3,   256, 256, 64,  128, 128, 4, 4, 2, 1, 64,   64,  0, F_OUTRELU);
  conv(pEb2, 4,  a1, a2, nullptr, nullptr, 64,  128, 128, 128, 64,  64,  4, 4, 2, 1, 1024, 128, 0, F_OUTRELU);
  conv(pEb3, 6,  a2, a3, nullptr, nullptr, 128, 64,  64,  128, 64,  64,  3, 3, 1, 1, 1152, 128, 0, 0);
  conv(pRE0a, 8,  a3,   rtmp, nullptr, nullptr, 128, 64, 64, 32,  64, 64, 3, 3, 1, 1, 1152, 32,  0, F_INRELU);
  conv(pRE0b, 10, rtmp, a2,   nullptr, a3,      32,  64, 64, 128, 64, 64, 1, 1, 1, 0, 32,   128, 0, F_INRELU);
  conv(pRE1a, 12, a2,   rtmp, nullptr, nullptr, 128, 64, 64, 32,  64, 64, 3, 3, 1, 1, 1152, 32,  0, F_INRELU);
  conv(pRE1b, 14, rtmp, a3,   nullptr, a2,      32,  64, 64, 128, 64, 64, 1, 1, 1, 0, 32,   128, 0, F_INRELU);
  // a3 = enc_b pre-relu

  // ---- encoder top: 64 -> 32 ----
  conv(pT1, 16, a3, b1, nullptr, nullptr, 128, 64, 64, 64,  32, 32, 4, 4, 2, 1, 2048, 64,  0, F_INRELU | F_OUTRELU);
  conv(pT2, 18, b1, b2, nullptr, nullptr, 64,  32, 32, 128, 32, 32, 3, 3, 1, 1, 576,  128, 0, 0);
  conv(pRT0a, 20, b2,    rtmpt, nullptr, nullptr, 128, 32, 32, 32,  32, 32, 3, 3, 1, 1, 1152, 32,  0, F_INRELU);
  conv(pRT0b, 22, rtmpt, b3,    nullptr, b2,      32,  32, 32, 128, 32, 32, 1, 1, 1, 0, 32,   128, 0, F_INRELU);
  conv(pRT1a, 24, b3,    rtmpt, nullptr, nullptr, 128, 32, 32, 32,  32, 32, 3, 3, 1, 1, 1152, 32,  0, F_INRELU);
  conv(pRT1b, 26, rtmpt, b2,    nullptr, b3,      32,  32, 32, 128, 32, 32, 1, 1, 1, 0, 32,   128, 0, F_INRELU);
  // b2 = enc_t pre-relu

  // ---- top quantization ----
  conv(pTop, 28, b2, zt, nullptr, nullptr, 128, 32, 32, 64, 32, 32, 1, 1, 1, 0, 128, 64, 0, F_INRELU);
  vq(zt, 29, qt, loss + 0, 32, 32, 64, 0);

  // ---- decoder top: 32 -> 64 (writes channels 0..63 of cat) ----
  conv(pDt1, 31, qt, b3, nullptr, nullptr, 64, 32, 32, 128, 32, 32, 3, 3, 1, 1, 576, 128, 0, 0);
  conv(pRD0a, 33, b3,    rtmpt, nullptr, nullptr, 128, 32, 32, 32,  32, 32, 3, 3, 1, 1, 1152, 32,  0, F_INRELU);
  conv(pRD0b, 35, rtmpt, b2,    nullptr, b3,      32,  32, 32, 128, 32, 32, 1, 1, 1, 0, 32,   128, 0, F_INRELU);
  conv(pRD1a, 37, b2,    rtmpt, nullptr, nullptr, 128, 32, 32, 32,  32, 32, 3, 3, 1, 1, 1152, 32,  0, F_INRELU);
  conv(pRD1b, 39, rtmpt, b3,    nullptr, b2,      32,  32, 32, 128, 32, 32, 1, 1, 1, 0, 32,   128, 0, F_INRELU);
  conv(pDtT, 41, b3, cat, nullptr, nullptr, 128, 32, 32, 64, 64, 64, 4, 4, 2, 1, 2048, 192, 0, F_INRELU | F_TRANS);

  // enc_b (relu) -> cat channels 64..191
  {
    int nel = NB * 128 * 64 * 64;
    copy_ch<<<(nel + 255) / 256, 256, 0, stream>>>(a3, cat, nel, 128, 64 * 64, 192, 64, 1);
  }

  // ---- bottom quantization ----
  conv(pBot, 43, cat, zb, nullptr, nullptr, 192, 64, 64, 64, 64, 64, 1, 1, 1, 0, 192, 64, 0, 0);
  vq(zb, 44, qcat, loss + 1, 64, 64, 128, 64);          // qb -> qcat ch 64..127
  conv(pUp, 46, qt, qcat, nullptr, nullptr, 64, 32, 32, 64, 64, 64, 4, 4, 2, 1, 1024, 128, 0, F_TRANS); // up -> ch 0..63

  // ---- final decoder: 64 -> 256 ----
  conv(pD1, 48, qcat, a2, nullptr, nullptr, 128, 64, 64, 128, 64, 64, 3, 3, 1, 1, 1152, 128, 0, 0);
  conv(pRX0a, 50, a2,   rtmp, nullptr, nullptr, 128, 64, 64, 32,  64, 64, 3, 3, 1, 1, 1152, 32,  0, F_INRELU);
  conv(pRX0b, 52, rtmp, a3,   nullptr, a2,      32,  64, 64, 128, 64, 64, 1, 1, 1, 0, 32,   128, 0, F_INRELU);
  conv(pRX1a, 54, a3,   rtmp, nullptr, nullptr, 128, 64, 64, 32,  64, 64, 3, 3, 1, 1, 1152, 32,  0, F_INRELU);
  conv(pRX1b, 56, rtmp, a2,   nullptr, a3,      32,  64, 64, 128, 64, 64, 1, 1, 1, 0, 32,   128, 0, F_INRELU);
  conv(pDT1, 58, a2, a1, nullptr, nullptr, 128, 64,  64,  64, 128, 128, 4, 4, 2, 1, 2048, 64, 0, F_INRELU | F_TRANS);
  conv(pDT2, 60, a1, nullptr, (float*)d_out, nullptr, 64, 128, 128, 3, 256, 256, 4, 4, 2, 1, 1024, 3, 0, F_INRELU | F_TRANS);

  // ---- losses ----
  const int NR = NB * 3 * 256 * 256;
  recon_sse<<<(NR + 255) / 256, 256, 0, stream>>>((const float*)d_out, xf, loss + 2, NR);
  finalize_total<<<1, 1, 0, stream>>>(
      loss, (float*)d_out + NR,
      1.f / (float)(NB * 32 * 32 * 64),   // top z elements
      1.f / (float)(NB * 64 * 64 * 64),   // bottom z elements
      1.f / (float)NR);
}
